// MambaFormer_57432302682154
// MI455X (gfx1250) — matrix operand order repaired and checked by reading, not compiled, not run on hardware
//
#include <hip/hip_runtime.h>

// ---------------------------------------------------------------------------
// MambaFormer block for MI455X (gfx1250, wave32, WMMA 16x16x32 f16).
// Pipeline:
//   [cvt weights+inputs -> f16] -> [WMMA gemm+silu] -> [dwconv17 (f32+f16 out)]
//   -> [WMMA gemm+sigmoid*x] -> [chunked SSM scan, 3 passes] -> [WMMA gemm+bias]
// GEMM A operands are pre-converted f16 (loaded as aligned v8h pairs, zero
// VALU converts in the hot loop); each wave computes a 16x32 tile (two WMMAs
// per K-step sharing one A fragment).
// ---------------------------------------------------------------------------

typedef __attribute__((ext_vector_type(16))) _Float16 v16h;
typedef __attribute__((ext_vector_type(8)))  _Float16 v8h;
typedef __attribute__((ext_vector_type(8)))  float    v8f;

#define D_MODEL 128
#define D_EXP   256
#define D_STATE 16
#define KCONV   17
#define SEQ     2048
#define BATCH   8
#define NTOK    (BATCH * SEQ)      // 16384 tokens
#define CHUNK   32                 // scan chunk length (even => A^CHUNK > 0)
#define NCHUNK  (SEQ / CHUNK)      // 64

// ---------------------------------------------------------------------------
// 16x32 tile: two V_WMMA_F32_16X16X32_F16 per K-step sharing the A fragment.
// A: f16 row-major [M][Kdim].  A fragment (this code's k order): per lane the 16 halves
//    are two contiguous runs of 8 (K = half*16..+7 and K = half*16+8..+7), same order as B.
// B: f16 row-major [N][Kdim] (W[e][d]); lane holds N=lane&15, K-half by lane>>4.
// ---------------------------------------------------------------------------
__device__ __forceinline__ void wmma_tile2(
    const _Float16* __restrict__ A16, int lda,
    const _Float16* __restrict__ B16, int ldb,
    int Kdim, int lane, v8f& acc0, v8f& acc1) {
    const int row  = lane & 15;
    const int half = lane >> 4;
    const _Float16* Arow = A16 + (size_t)row * lda + half * 16;
    const _Float16* B0   = B16 + (size_t)row * ldb + half * 16;
    const _Float16* B1   = B0  + (size_t)16 * ldb;
    for (int k0 = 0; k0 < Kdim; k0 += 32) {
        const v8h lo = *(const v8h*)(Arow + k0);        // K = half*16 .. +7
        const v8h hi = *(const v8h*)(Arow + k0 + 8);    // K = half*16+8 .. +7
        v16h a;
#pragma unroll
        for (int i = 0; i < 8; ++i) { a[i] = lo[i]; a[8 + i] = hi[i]; }
        const v16h b0 = *(const v16h*)(B0 + k0);
        const v16h b1 = *(const v16h*)(B1 + k0);
        acc0 = __builtin_amdgcn_wmma_f32_16x16x32_f16(false, a, false, b0,
                                                      (short)0, acc0, false, false);
        acc1 = __builtin_amdgcn_wmma_f32_16x16x32_f16(false, a, false, b1,
                                                      (short)0, acc1, false, false);
    }
}

// ---------------------------------------------------------------------------
// Kernel 0a: weights -> f16.
// ---------------------------------------------------------------------------
__global__ __launch_bounds__(256) void convert_weights(
    const float* __restrict__ We, const float* __restrict__ Wg,
    const float* __restrict__ Wo,
    _Float16* __restrict__ We16, _Float16* __restrict__ Wg16,
    _Float16* __restrict__ Wo16) {
    const int i = blockIdx.x * 256 + threadIdx.x;
    if (i < D_EXP * D_MODEL)  We16[i] = (_Float16)We[i];
    if (i < D_EXP * D_EXP)    Wg16[i] = (_Float16)Wg[i];
    if (i < D_MODEL * D_EXP)  Wo16[i] = (_Float16)Wo[i];
}

// Kernel 0b: inputs -> f16 (one float2 per thread).
__global__ __launch_bounds__(256) void convert_inputs(
    const float* __restrict__ X, _Float16* __restrict__ X16, int n2) {
    const int i = blockIdx.x * 256 + threadIdx.x;
    if (i < n2) {
        const float2 f = *(const float2*)(X + 2 * (size_t)i);
        X16[2 * (size_t)i]     = (_Float16)f.x;
        X16[2 * (size_t)i + 1] = (_Float16)f.y;
    }
}

// ---------------------------------------------------------------------------
// Kernel 1: x_exp = silu(inputs @ W_exp^T + b_exp).  M=NTOK, N=256, K=128.
// One wave per 16x32 tile.
// ---------------------------------------------------------------------------
__global__ __launch_bounds__(256) void gemm_exp_silu(
    const _Float16* __restrict__ X16, const _Float16* __restrict__ W16,
    const float* __restrict__ bias, float* __restrict__ Y) {
    const int wave = (blockIdx.x * 256 + threadIdx.x) >> 5;
    const int lane = threadIdx.x & 31;
    const int tilesN2 = D_EXP / 32;                // 8
    const int m0 = (wave / tilesN2) * 16;
    const int np = (wave % tilesN2) * 32;

    v8f acc0 = {}, acc1 = {};
    wmma_tile2(X16 + (size_t)m0 * D_MODEL, D_MODEL,
               W16 + (size_t)np * D_MODEL, D_MODEL, D_MODEL, lane, acc0, acc1);

    const int row  = lane & 15;
    const int half = lane >> 4;
#pragma unroll
    for (int t = 0; t < 2; ++t) {
        const int n    = np + t * 16 + row;
        const float bn = bias[n];
        const v8f&  acc = t ? acc1 : acc0;
#pragma unroll
        for (int r = 0; r < 8; ++r) {
            const int m   = m0 + r + half * 8;
            const float v = acc[r] + bn;
            Y[(size_t)m * D_EXP + n] = v / (1.0f + __expf(-v));  // silu
        }
    }
}

// ---------------------------------------------------------------------------
// Kernel 2: depthwise conv K=17 SAME along L; writes f32 (for gating) and
// f16 (gate-GEMM A operand).
// ---------------------------------------------------------------------------
__global__ __launch_bounds__(256) void dwconv17(
    const float* __restrict__ X, const float* __restrict__ w,
    const float* __restrict__ bias, float* __restrict__ Y32,
    _Float16* __restrict__ Y16) {
    const int e = threadIdx.x;
    const int t = blockIdx.x;
    const int b = t / SEQ;
    const int l = t % SEQ;
    float acc = bias[e];
#pragma unroll
    for (int k = 0; k < KCONV; ++k) {
        const int ls = l + k - (KCONV / 2);
        if (ls >= 0 && ls < SEQ)
            acc = fmaf(X[((size_t)b * SEQ + ls) * D_EXP + e],
                       w[e * KCONV + k], acc);
    }
    Y32[(size_t)t * D_EXP + e] = acc;
    Y16[(size_t)t * D_EXP + e] = (_Float16)acc;
}

// ---------------------------------------------------------------------------
// Kernel 3: gate = sigmoid(x_conv @ W_gate^T + b_gate); x_gated = x_conv*gate.
// M=NTOK, N=256, K=256.  A operand from the f16 copy; gating uses f32 copy.
// ---------------------------------------------------------------------------
__global__ __launch_bounds__(256) void gemm_gate_mul(
    const _Float16* __restrict__ XC16, const float* __restrict__ XC32,
    const _Float16* __restrict__ W16, const float* __restrict__ bias,
    float* __restrict__ XG) {
    const int wave = (blockIdx.x * 256 + threadIdx.x) >> 5;
    const int lane = threadIdx.x & 31;
    const int tilesN2 = D_EXP / 32;                // 8
    const int m0 = (wave / tilesN2) * 16;
    const int np = (wave % tilesN2) * 32;

    v8f acc0 = {}, acc1 = {};
    wmma_tile2(XC16 + (size_t)m0 * D_EXP, D_EXP,
               W16 + (size_t)np * D_EXP, D_EXP, D_EXP, lane, acc0, acc1);

    const int row  = lane & 15;
    const int half = lane >> 4;
#pragma unroll
    for (int t = 0; t < 2; ++t) {
        const int n    = np + t * 16 + row;
        const float bn = bias[n];
        const v8f&  acc = t ? acc1 : acc0;
#pragma unroll
        for (int r = 0; r < 8; ++r) {
            const int m    = m0 + r + half * 8;
            const float g  = 1.0f / (1.0f + __expf(-(acc[r] + bn)));
            const size_t o = (size_t)m * D_EXP + n;
            XG[o] = XC32[o] * g;
        }
    }
}

// ---------------------------------------------------------------------------
// SSM scan, chunked 3-pass (exact: A is diagonal, recurrence is linear).
//   h_end(chunk, true) = A^CHUNK (.) h_start + h_end(chunk, zero-init)
// Pass 1: per (b,e,chunk) local scan from h=0, store chunk-final h.
// Pass 2: per (b,e,s)   compose carries across 64 chunks (sequential, tiny).
// Pass 3: per (b,e,chunk) replay from carry, emit y as f16.
// ---------------------------------------------------------------------------
__global__ __launch_bounds__(256) void ssm_pass1(
    const float* __restrict__ XG, const float* __restrict__ A_log,
    const float* __restrict__ Bm, float* __restrict__ hend) {
    const int e = threadIdx.x;
    const int b = blockIdx.x / NCHUNK;
    const int c = blockIdx.x % NCHUNK;

    float A[D_STATE], Bl[D_STATE], h[D_STATE];
#pragma unroll
    for (int s = 0; s < D_STATE; ++s) {
        A[s]  = -__expf(A_log[e * D_STATE + s]);
        Bl[s] = Bm[e * D_STATE + s];
        h[s]  = 0.0f;
    }
    const float* xg = XG + ((size_t)b * SEQ + c * CHUNK) * D_EXP + e;
    for (int l = 0; l < CHUNK; ++l) {
        const float u = xg[(size_t)l * D_EXP];
#pragma unroll
        for (int s = 0; s < D_STATE; ++s)
            h[s] = fmaf(A[s], h[s], u * Bl[s]);
    }
    float* he = hend + ((((size_t)b * NCHUNK + c) * D_EXP + e) << 4);
#pragma unroll
    for (int s = 0; s < D_STATE; ++s) he[s] = h[s];
}

__global__ __launch_bounds__(256) void ssm_pass2(
    const float* __restrict__ A_log, const float* __restrict__ hend,
    float* __restrict__ carry) {
    const int idx = blockIdx.x * 256 + threadIdx.x;   // 0..32767
    const int b   = idx >> 12;                        // (e,s) = 4096 per batch
    const int es  = idx & 4095;
    const float Ap = __expf((float)CHUNK * A_log[es]); // (-exp(aL))^32, 32 even
    float cr = 0.0f;
    for (int c = 0; c < NCHUNK; ++c) {
        const size_t o = ((((size_t)b * NCHUNK + c)) << 12) + es;
        carry[o] = cr;                    // h at the start of chunk c
        cr = fmaf(Ap, cr, hend[o]);
    }
}

__global__ __launch_bounds__(256) void ssm_pass3(
    const float* __restrict__ XG, const float* __restrict__ A_log,
    const float* __restrict__ Bm, const float* __restrict__ Cm,
    const float* __restrict__ Dv, const float* __restrict__ carry,
    _Float16* __restrict__ Y16) {
    const int e = threadIdx.x;
    const int b = blockIdx.x / NCHUNK;
    const int c = blockIdx.x % NCHUNK;

    float A[D_STATE], Bl[D_STATE], Cl[D_STATE], h[D_STATE];
    const float* cr = carry + ((((size_t)b * NCHUNK + c) * D_EXP + e) << 4);
#pragma unroll
    for (int s = 0; s < D_STATE; ++s) {
        A[s]  = -__expf(A_log[e * D_STATE + s]);
        Bl[s] = Bm[e * D_STATE + s];
        Cl[s] = Cm[e * D_STATE + s];
        h[s]  = cr[s];
    }
    const float d = Dv[e];
    const float* xg = XG + ((size_t)b * SEQ + c * CHUNK) * D_EXP + e;
    _Float16*    y  = Y16 + ((size_t)b * SEQ + c * CHUNK) * D_EXP + e;
    for (int l = 0; l < CHUNK; ++l) {
        const float u = xg[(size_t)l * D_EXP];
        float ac[4] = {d * u, 0.0f, 0.0f, 0.0f};     // 4-way split accumulator
#pragma unroll
        for (int s = 0; s < D_STATE; ++s) {
            h[s] = fmaf(A[s], h[s], u * Bl[s]);
            ac[s & 3] = fmaf(Cl[s], h[s], ac[s & 3]);
        }
        y[(size_t)l * D_EXP] = (_Float16)((ac[0] + ac[1]) + (ac[2] + ac[3]));
    }
}

// ---------------------------------------------------------------------------
// Kernel 5: out = y @ W_out^T + b_out.  M=NTOK, N=128, K=256.
// ---------------------------------------------------------------------------
__global__ __launch_bounds__(256) void gemm_out(
    const _Float16* __restrict__ Y16, const _Float16* __restrict__ W16,
    const float* __restrict__ bias, float* __restrict__ Out) {
    const int wave = (blockIdx.x * 256 + threadIdx.x) >> 5;
    const int lane = threadIdx.x & 31;
    const int tilesN2 = D_MODEL / 32;              // 4
    const int m0 = (wave / tilesN2) * 16;
    const int np = (wave % tilesN2) * 32;

    v8f acc0 = {}, acc1 = {};
    wmma_tile2(Y16 + (size_t)m0 * D_EXP, D_EXP,
               W16 + (size_t)np * D_EXP, D_EXP, D_EXP, lane, acc0, acc1);

    const int row  = lane & 15;
    const int half = lane >> 4;
#pragma unroll
    for (int t = 0; t < 2; ++t) {
        const int n    = np + t * 16 + row;
        const float bn = bias[n];
        const v8f&  acc = t ? acc1 : acc0;
#pragma unroll
        for (int r = 0; r < 8; ++r) {
            const int m = m0 + r + half * 8;
            Out[(size_t)m * D_MODEL + n] = acc[r] + bn;
        }
    }
}

// ---------------------------------------------------------------------------
// Host-side launcher.
// Input order: inputs, W_exp, b_exp, conv_w, conv_b, A_log, B, C, D,
//              W_gate, b_gate, W_out, b_out
// ---------------------------------------------------------------------------
extern "C" void kernel_launch(void* const* d_in, const int* in_sizes, int n_in,
                              void* d_out, int out_size, void* d_ws, size_t ws_size,
                              hipStream_t stream) {
    const float* inputs = (const float*)d_in[0];
    const float* W_exp  = (const float*)d_in[1];
    const float* b_exp  = (const float*)d_in[2];
    const float* conv_w = (const float*)d_in[3];
    const float* conv_b = (const float*)d_in[4];
    const float* A_log  = (const float*)d_in[5];
    const float* Bmat   = (const float*)d_in[6];
    const float* Cmat   = (const float*)d_in[7];
    const float* Dvec   = (const float*)d_in[8];
    const float* W_gate = (const float*)d_in[9];
    const float* b_gate = (const float*)d_in[10];
    const float* W_out  = (const float*)d_in[11];
    const float* b_out  = (const float*)d_in[12];
    float* out = (float*)d_out;

    // ---- workspace carving (regions 256-B aligned) ----
    char* ws = (char*)d_ws;
    size_t off = 0;
    _Float16* We16 = (_Float16*)(ws + off); off += (size_t)D_EXP * D_MODEL * 2;   //  64 KB
    _Float16* Wg16 = (_Float16*)(ws + off); off += (size_t)D_EXP * D_EXP * 2;     // 128 KB
    _Float16* Wo16 = (_Float16*)(ws + off); off += (size_t)D_MODEL * D_EXP * 2;   //  64 KB
    _Float16* in16 = (_Float16*)(ws + off); off += (size_t)NTOK * D_MODEL * 2;    //   4 MB
    _Float16* xc16 = (_Float16*)(ws + off); off += (size_t)NTOK * D_EXP * 2;      //   8 MB
    _Float16* y16  = (_Float16*)(ws + off); off += (size_t)NTOK * D_EXP * 2;      //   8 MB
    float* x_exp   = (float*)(ws + off);    off += (size_t)NTOK * D_EXP * 4;      //  16 MB
    float* xc32    = (float*)(ws + off);    off += (size_t)NTOK * D_EXP * 4;      //  16 MB
    float* xg32    = (float*)(ws + off);    off += (size_t)NTOK * D_EXP * 4;      //  16 MB
    // x_exp is dead after dwconv17: reuse its 16 MB for hend (8 MB) + carry (8 MB)
    float* hend  = x_exp;
    float* carry = x_exp + (size_t)BATCH * NCHUNK * D_EXP * D_STATE;

    // 0) converts
    convert_weights<<<(D_EXP * D_EXP + 255) / 256, 256, 0, stream>>>(
        W_exp, W_gate, W_out, We16, Wg16, Wo16);
    convert_inputs<<<(NTOK * D_MODEL / 2 + 255) / 256, 256, 0, stream>>>(
        inputs, in16, NTOK * D_MODEL / 2);

    // 1) x_exp = silu(inputs @ W_exp^T + b_exp)
    gemm_exp_silu<<<(NTOK / 16) * (D_EXP / 32) / 8, 256, 0, stream>>>(
        in16, We16, b_exp, x_exp);
    // 2) depthwise conv
    dwconv17<<<NTOK, 256, 0, stream>>>(x_exp, conv_w, conv_b, xc32, xc16);
    // 3) gating
    gemm_gate_mul<<<(NTOK / 16) * (D_EXP / 32) / 8, 256, 0, stream>>>(
        xc16, xc32, Wg16, b_gate, xg32);
    // 4) chunked SSM scan
    ssm_pass1<<<BATCH * NCHUNK, 256, 0, stream>>>(xg32, A_log, Bmat, hend);
    ssm_pass2<<<BATCH * D_EXP * D_STATE / 256, 256, 0, stream>>>(A_log, hend, carry);
    ssm_pass3<<<BATCH * NCHUNK, 256, 0, stream>>>(xg32, A_log, Bmat, Cmat, Dvec,
                                                  carry, y16);
    // 5) out = y @ W_out^T + b_out
    gemm_out<<<(NTOK / 16) * (D_MODEL / 32) / 8, 256, 0, stream>>>(
        y16, Wo16, b_out, out);
}
